// BoundaryPredictor2_76742475644943
// MI455X (gfx1250) — compile-verified, matrix operand order checked by reading, not run on hardware
//
#include <hip/hip_runtime.h>
#include <hip/hip_bf16.h>

#define BB 8
#define LL 1024
#define DD 1024
#define TILE 16
#define STR 1032   // LDS row stride in bf16 elements (2064 B, 16B aligned)

typedef __attribute__((ext_vector_type(16))) __bf16 v16bf;
typedef __attribute__((ext_vector_type(8)))  __bf16 v8bf;
typedef __attribute__((ext_vector_type(4)))  __bf16 v4bf;
typedef __attribute__((ext_vector_type(8)))  float  v8f;

// ---------------- K0: zero pooled output + ws scalars ----------------
__global__ void zero_kernel(float* __restrict__ out, int n, float* __restrict__ scal) {
    int i = blockIdx.x * 256 + threadIdx.x;
    if (i < n) out[i] = 0.0f;
    if (blockIdx.x == 0 && threadIdx.x < 16) scal[threadIdx.x] = 0.0f;
}

// ---------------- K1: fused row-norm + normalize + bf16 convert ----------------
// One wave per row; the 32 f32/lane stay in registers across the reduction.
__global__ __launch_bounds__(256) void norm_convert_kernel(const float* __restrict__ hidden,
                                                           __bf16* __restrict__ nbf) {
    int wave = threadIdx.x >> 5;
    int lane = threadIdx.x & 31;
    int row  = blockIdx.x * 8 + wave;                 // 8192 rows total
    const float4* src = (const float4*)(hidden + (size_t)row * DD);
    float4 v[8];
    float ss = 0.0f;
#pragma unroll
    for (int i = 0; i < 8; ++i) {
        v[i] = src[lane + i * 32];
        ss += v[i].x * v[i].x + v[i].y * v[i].y + v[i].z * v[i].z + v[i].w * v[i].w;
    }
    ss += __shfl_xor(ss, 1, 32);
    ss += __shfl_xor(ss, 2, 32);
    ss += __shfl_xor(ss, 4, 32);
    ss += __shfl_xor(ss, 8, 32);
    ss += __shfl_xor(ss, 16, 32);
    float inv = 1.0f / fmaxf(sqrtf(ss), 1e-12f);
#pragma unroll
    for (int i = 0; i < 8; ++i) {
        v4bf o;
        o[0] = (__bf16)(v[i].x * inv);
        o[1] = (__bf16)(v[i].y * inv);
        o[2] = (__bf16)(v[i].z * inv);
        o[3] = (__bf16)(v[i].w * inv);
        *(v4bf*)(nbf + (size_t)row * DD + (size_t)(lane + i * 32) * 4) = o;
    }
}

// ---------------- K2: convert Wq/Wk to bf16 ----------------
__global__ __launch_bounds__(256) void convert_w(const float* __restrict__ wq,
                                                 const float* __restrict__ wk,
                                                 __bf16* __restrict__ wqb,
                                                 __bf16* __restrict__ wkb) {
    int i = blockIdx.x * 256 + threadIdx.x;           // D*D = 1M threads
    wqb[i] = (__bf16)wq[i];
    wkb[i] = (__bf16)wk[i];
}

// ---------------- helpers for WMMA fragments ----------------
__device__ inline v16bf mk16(const __bf16* p0, const __bf16* p1) {
    v8bf lo = *(const v8bf*)p0;
    v8bf hi = *(const v8bf*)p1;
    v16bf r;
#pragma unroll
    for (int i = 0; i < 8; ++i) { r[i] = lo[i]; r[i + 8] = hi[i]; }
    return r;
}

// ---------------- K3: fused (n @ Wq^T) . (n' @ Wk^T) via WMMA ----------------
// One wave handles 16 consecutive positions of one batch. LDS staged with
// CDNA5 async global->LDS DMA (ASYNCcnt), no VGPR round-trip.
__global__ __launch_bounds__(32) void cos_wmma_kernel(const __bf16* __restrict__ nbf,
                                                      const __bf16* __restrict__ wq,
                                                      const __bf16* __restrict__ wk,
                                                      float* __restrict__ probs) {
    __shared__ __bf16 lds[17 * STR];
    int l0   = blockIdx.x * TILE;                     // 64 tiles
    int b    = blockIdx.y;                            // 8 batches
    int lane = threadIdx.x;

    // LDS byte offset of the staging buffer (low 32 bits of the flat address).
    unsigned ldsbase = (unsigned)(size_t)&lds[0];

    // Async-stage 17 normalized bf16 rows (16B chunks, 128 per row).
    for (int t = lane; t < 17 * 128; t += 32) {
        int r = t >> 7;
        int c = t & 127;
        int l = l0 + r; if (l > LL - 1) l = LL - 1;
        const __bf16* g = nbf + (size_t)(b * LL + l) * DD + (size_t)c * 8;
        unsigned laddr = ldsbase + (unsigned)(r * (STR * 2) + c * 16);
        asm volatile("global_load_async_to_lds_b128 %0, %1, off"
                     :: "v"(laddr), "v"(g)
                     : "memory");
    }
    asm volatile("s_wait_asynccnt 0x0" ::: "memory");
    __syncthreads();

    int m    = lane & 15;                             // A-matrix row / B-matrix column
    int hh   = lane >> 4;                             // lane half
    int koff = hh * 8;                                // K sub-offset per 16-bit layout

    v8f coscc = {};
    for (int nc = 0; nc < DD / 16; ++nc) {            // 64 output-column tiles
        v8f accq = {};
        v8f acck = {};
        const __bf16* wqrow = wq + (size_t)(nc * 16 + m) * DD;
        const __bf16* wkrow = wk + (size_t)(nc * 16 + m) * DD;
        if (nc + 1 < DD / 16) {
            __builtin_prefetch(wq + (size_t)(nc * 16 + 16 + m) * DD, 0, 1);
            __builtin_prefetch(wk + (size_t)(nc * 16 + 16 + m) * DD, 0, 1);
        }
#pragma unroll 4
        for (int kk = 0; kk < DD / 32; ++kk) {        // full K reduction
            int k0 = kk * 32;
            // A fragments (16-bit 16x32 layout) from LDS
            v16bf aq = mk16(&lds[m * STR + k0 + koff],       &lds[m * STR + k0 + koff + 16]);
            v16bf ak = mk16(&lds[(m + 1) * STR + k0 + koff], &lds[(m + 1) * STR + k0 + koff + 16]);
            // B fragments: contiguous 16B chunks of W rows (W row-major [n][k])
            v16bf bq = mk16(wqrow + k0 + koff, wqrow + k0 + koff + 16);
            v16bf bk = mk16(wkrow + k0 + koff, wkrow + k0 + koff + 16);
            accq = __builtin_amdgcn_wmma_f32_16x16x32_bf16(false, aq, false, bq,
                                                           (short)0, accq, false, false);
            acck = __builtin_amdgcn_wmma_f32_16x16x32_bf16(false, ak, false, bk,
                                                           (short)0, acck, false, false);
        }
#pragma unroll
        for (int r = 0; r < 8; ++r) coscc[r] += accq[r] * acck[r];  // row-wise Q.K partials
    }

    // Reduce over the 16 lanes of each half (columns) -> cos per position.
#pragma unroll
    for (int r = 0; r < 8; ++r) {
        float v = coscc[r];
        v += __shfl_xor(v, 1, 32);
        v += __shfl_xor(v, 2, 32);
        v += __shfl_xor(v, 4, 32);
        v += __shfl_xor(v, 8, 32);
        coscc[r] = v;
    }
    if ((lane & 15) == 0) {
#pragma unroll
        for (int r = 0; r < 8; ++r) {
            int p = l0 + hh * 8 + r;                  // cos position (q row index)
            if (p < LL - 1) {
                float pr = (1.0f - coscc[r]) * 0.5f;
                pr = fminf(fmaxf(pr, 0.0f), 1.0f);
                probs[b * LL + p + 1] = pr;           // probs[.,0] synthesized later
            }
        }
    }
}

// ---------------- K4: boundaries, prefix-scan seg ids, counts, short mask ----------------
__global__ __launch_bounds__(1024) void boundary_kernel(const float* __restrict__ probs,
                                                        const float* __restrict__ noise,
                                                        const float* __restrict__ mask,
                                                        int* __restrict__ segid,
                                                        float* __restrict__ scal,
                                                        float* __restrict__ smask_out) {
    __shared__ float s[LL];
    int b = blockIdx.x;
    int l = threadIdx.x;
    const float tiny = 1.17549435e-38f;

    float pr = (l == 0) ? 1.0f : probs[b * LL + l];
    float p  = fminf(fmaxf(pr, tiny), 1.0f - tiny);
    float u  = fminf(fmaxf(noise[b * LL + l], tiny), 1.0f - tiny);
    float logit = logf(p) - log1pf(-p) + logf(u) - log1pf(-u);   // TEMP = 1
    float soft  = 1.0f / (1.0f + expf(-logit));
    float hard  = (soft > 0.5f) ? 1.0f : 0.0f;
    float mk    = mask[b * LL + l];
    hard *= mk;

    s[l] = hard;
    __syncthreads();
    // inclusive Hillis-Steele scan
    for (int off = 1; off < LL; off <<= 1) {
        float v = (l >= off) ? s[l - off] : 0.0f;
        __syncthreads();
        s[l] += v;
        __syncthreads();
    }
    float incl = s[l];
    segid[b * LL + l] = (int)(incl - hard);           // exclusive prefix = segment id
    float cnt = s[LL - 1];                            // boundaries in this batch
    smask_out[b * LL + l] = (l < (int)cnt) ? 1.0f : 0.0f;

    __syncthreads();
    s[l] = mk;                                        // mask sum (total positions)
    __syncthreads();
    for (int off = 512; off > 0; off >>= 1) {
        if (l < off) s[l] += s[l + off];
        __syncthreads();
    }
    if (l == 0) {
        atomicAdd(&scal[0], cnt);                     // num_boundaries
        atomicAdd(&scal[1], s[0]);                    // total_positions
    }
}

// ---------------- K5: binomial loss scalar ----------------
__global__ void loss_kernel(const float* __restrict__ scal, float* __restrict__ out3) {
    if (threadIdx.x == 0 && blockIdx.x == 0) {
        float n = scal[0], T = scal[1];
        float logp = lgammaf(T + 1.0f) - lgammaf(n + 1.0f) - lgammaf(T - n + 1.0f)
                   + n * logf(0.2f) + (T - n) * log1pf(-0.2f);
        out3[0] = -logp / T;                          // loss
        out3[1] = n;                                  // num_boundaries
        out3[2] = T;                                  // total_positions
    }
}

// ---------------- K6: segmented mean-pool ----------------
__global__ __launch_bounds__(256) void pool_kernel(const float* __restrict__ hidden,
                                                   const int* __restrict__ segid,
                                                   float* __restrict__ pooled) {
    int b = blockIdx.y;
    int d = blockIdx.x * 256 + threadIdx.x;
    float acc = 0.0f, cnt = 0.0f;
    int seg = 0;
    for (int l = 0; l < LL; ++l) {
        int sg = segid[b * LL + l];                   // broadcast load per block
        if (sg != seg) {
            pooled[((size_t)b * LL + seg) * DD + d] = acc / (cnt + 1e-9f);
            acc = 0.0f; cnt = 0.0f; seg = sg;
        }
        acc += hidden[((size_t)b * LL + l) * DD + d];
        cnt += 1.0f;
    }
    pooled[((size_t)b * LL + seg) * DD + d] = acc / (cnt + 1e-9f);
}

// ---------------- launch ----------------
extern "C" void kernel_launch(void* const* d_in, const int* in_sizes, int n_in,
                              void* d_out, int out_size, void* d_ws, size_t ws_size,
                              hipStream_t stream) {
    const float* hidden = (const float*)d_in[0];
    const float* mask   = (const float*)d_in[1];
    const float* noise  = (const float*)d_in[2];
    const float* Wq     = (const float*)d_in[3];
    const float* Wk     = (const float*)d_in[4];
    float* out = (float*)d_out;

    char* ws = (char*)d_ws;
    float*  probs = (float*)(ws);                          // 32 KB
    int*    segid = (int*)  (ws + 32768);                  // 32 KB
    float*  scal  = (float*)(ws + 65536);                  // 64 B
    __bf16* wqb   = (__bf16*)(ws + 131072);                // 2 MB
    __bf16* wkb   = (__bf16*)(ws + 131072 + 2097152);      // 2 MB
    __bf16* nbf   = (__bf16*)(ws + 131072 + 4194304);      // 16 MB normalized bf16 hidden

    const int nPooled = BB * LL * DD;                      // 8388608
    float* scalars_out = out + nPooled;                    // loss, nb, tp
    float* smask_out   = out + nPooled + 3;                // [B, L]

    zero_kernel<<<(nPooled + 255) / 256, 256, 0, stream>>>(out, nPooled, scal);
    norm_convert_kernel<<<(BB * LL) / 8, 256, 0, stream>>>(hidden, nbf);
    convert_w<<<(DD * DD) / 256, 256, 0, stream>>>(Wq, Wk, wqb, wkb);

    dim3 g3(LL / TILE, BB);                                // 64 x 8 tiles, 1 wave each
    cos_wmma_kernel<<<g3, 32, 0, stream>>>(nbf, wqb, wkb, probs);

    boundary_kernel<<<BB, LL, 0, stream>>>(probs, noise, mask, segid, scal, smask_out);
    loss_kernel<<<1, 32, 0, stream>>>(scal, scalars_out);

    dim3 g6(DD / 256, BB);
    pool_kernel<<<g6, 256, 0, stream>>>(hidden, segid, out);
}
